// AugmentedCTCLoss_48249662603499
// MI455X (gfx1250) — matrix operand order repaired and checked by reading, not compiled, not run on hardware
//
#include <hip/hip_runtime.h>
#include <hip/hip_bf16.h>
#include <math.h>

typedef __attribute__((ext_vector_type(16))) _Float16 v16h;
typedef __attribute__((ext_vector_type(8)))  float    v8f;

#define B_   64
#define T_   1024
#define V_   256
#define S_   256
#define D_   24
#define L_   513
#define NEGV (-1e10f)
#define BIGV (1e10f)
#define GAMMA_ 0.1f

// -------------------------------------------------------------------------
// Kernel 1: pred[B*T, D] = exp(log_probs)[B*T, V] @ feature_matrix[V, D]
// f16 WMMA (16x16x32), f32 accumulate. Each wave owns one 16-row M tile and
// two 16-col N tiles (cols 24..31 are zero padding in LDS-staged B).
// -------------------------------------------------------------------------
__global__ __launch_bounds__(256) void k_gemm(const float* __restrict__ lp,
                                              const float* __restrict__ fm,
                                              float* __restrict__ pred) {
    __shared__ _Float16 ldsB[V_ * 32];   // 16 KB, [v][n] with n padded to 32
    const int tid = threadIdx.x;
    {   // stage B: 256 threads, one V-row each
        const int v = tid;
        #pragma unroll
        for (int d = 0; d < D_; ++d) ldsB[v * 32 + d] = (_Float16)fm[v * D_ + d];
        #pragma unroll
        for (int d = D_; d < 32; ++d) ldsB[v * 32 + d] = (_Float16)0.0f;
    }
    __syncthreads();

    const int wave  = tid >> 5;
    const int lane  = tid & 31;
    const int mtile = blockIdx.x * 8 + wave;          // 512 blocks * 8 waves = 4096 tiles
    const int row   = mtile * 16 + (lane & 15);       // A-matrix M for this lane
    const int khiA  = (lane >> 4) * 8;                // lanes 16-31: K shifted by 8
    const int khiB  = (lane >> 4) * 16;               // lanes 16-31: K shifted by 16
    const int ncol  = lane & 15;                      // B-matrix N (within tile)

    v8f c0 = {}; v8f c1 = {};
    const float* lpr = lp + (size_t)row * V_;

    for (int kk = 0; kk < V_; kk += 32) {
        // A tile: halves 0..7 <- K = kk+khiA+0..7 ; halves 8..15 <- K = kk+khiA+16..23
        float buf[16];
        const float* p0 = lpr + kk + khiA;
        #pragma unroll
        for (int i = 0; i < 8; ++i) buf[i]     = p0[i];
        #pragma unroll
        for (int i = 0; i < 8; ++i) buf[8 + i] = p0[16 + i];
        v16h a;
        #pragma unroll
        for (int i = 0; i < 16; ++i) a[i] = (_Float16)__expf(buf[i]);

        // B tiles from LDS: half h <- K = kk+khiA+(h&7)+16*(h>>3), col = ncol (+16 for tile 1)
        v16h b0, b1;
        #pragma unroll
        for (int h = 0; h < 16; ++h) {
            const int vrow = kk + khiA + (h & 7) + ((h >> 3) << 4);
            b0[h] = ldsB[vrow * 32 + ncol];
            b1[h] = ldsB[vrow * 32 + 16 + ncol];
        }
        c0 = __builtin_amdgcn_wmma_f32_16x16x32_f16(false, a, false, b0, (short)0, c0, false, false);
        c1 = __builtin_amdgcn_wmma_f32_16x16x32_f16(false, a, false, b1, (short)0, c1, false, false);
    }

    // C/D layout: VGPR r -> M = r (+8 for lanes 16-31), N = lane&15
    const int rbase = mtile * 16 + ((lane >> 4) ? 8 : 0);
    #pragma unroll
    for (int r = 0; r < 8; ++r) {
        const int orow = rbase + r;
        pred[(size_t)orow * D_ + ncol] = c0[r];
        const int cgl = 16 + ncol;
        if (cgl < D_) pred[(size_t)orow * D_ + cgl] = c1[r];
    }
}

// -------------------------------------------------------------------------
// Kernel 2: CTC alpha recursion. One block per batch, 512 threads over L=513
// states, LDS ping-pong alpha buffers, one barrier per time step.
// -------------------------------------------------------------------------
__global__ __launch_bounds__(512) void k_ctc(const float* __restrict__ lp,
                                             const int* __restrict__ targets,
                                             const int* __restrict__ in_len,
                                             const int* __restrict__ tgt_len,
                                             float* __restrict__ out_per_b) {
    __shared__ int   ext[L_];
    __shared__ int   skip[L_];
    __shared__ float bufA[L_];
    __shared__ float bufB[L_];
    const int b   = blockIdx.x;
    const int tid = threadIdx.x;
    const float* lpb = lp + (size_t)b * T_ * V_;

    for (int s = tid; s < L_; s += 512)
        ext[s] = (s & 1) ? targets[b * S_ + (s >> 1)] : 0;
    __syncthreads();
    for (int s = tid; s < L_; s += 512) {
        skip[s] = (s >= 2 && ext[s] != 0 && ext[s] != ext[s - 2]) ? 1 : 0;
        bufA[s] = (s < 2) ? lpb[ext[s]] : NEGV;   // alpha0 with t=0 emission
    }
    __syncthreads();

    const int len = in_len[b];
    float* cur = bufA;
    float* nxt = bufB;
    for (int t = 1; t < len; ++t) {               // frozen past len: stop there
        const float* lpt = lpb + (size_t)t * V_;
        for (int s = tid; s < L_; s += 512) {
            const float a1 = cur[s];
            const float a2 = (s >= 1) ? cur[s - 1] : NEGV;
            const float a3 = skip[s] ? cur[s - 2] : NEGV;
            const float m  = fmaxf(fmaxf(a1, a2), a3);
            const float lse = m + __logf(__expf(a1 - m) + __expf(a2 - m) + __expf(a3 - m));
            nxt[s] = lse + lpt[ext[s]];
        }
        __syncthreads();
        float* tmp = cur; cur = nxt; nxt = tmp;
    }
    if (tid == 0) {
        const int tl = tgt_len[b];
        const int l  = 2 * tl + 1;
        const float al = cur[l - 1];
        const float ap = cur[l - 2];
        const float m  = fmaxf(al, ap);
        const float ll = m + __logf(__expf(al - m) + __expf(ap - m));
        out_per_b[b] = -ll / (float)tl;
    }
}

// -------------------------------------------------------------------------
// Kernel 3: soft-DTW anti-diagonal wavefront. One block per batch, 256
// threads (one per target column), 3 rotating diagonal buffers in LDS,
// squared-L2 cost computed on the fly (target features kept in registers).
// -------------------------------------------------------------------------
__global__ __launch_bounds__(256) void k_sdtw(const float* __restrict__ pred,
                                              const float* __restrict__ fm,
                                              const int* __restrict__ targets,
                                              float* __restrict__ out_per_b) {
    __shared__ float diag[3][S_ + 1];
    const int b   = blockIdx.x;
    const int tid = threadIdx.x;   // 0..255
    const int j   = tid + 1;       // DP column 1..256

    float tg[D_];
    float sqT = 0.f;
    {
        const int ti = targets[b * S_ + tid];
        #pragma unroll
        for (int d = 0; d < D_; ++d) { tg[d] = fm[ti * D_ + d]; sqT += tg[d] * tg[d]; }
    }

    // diag[0] = anti-diagonal k=0 (R[0][0]=0), diag[1] = k=1 (all BIG boundary)
    if (tid == 0) { diag[0][0] = 0.f; diag[1][0] = BIGV; diag[2][0] = BIGV; }
    diag[0][j] = BIGV;
    diag[1][j] = BIGV;
    __syncthreads();

    int p2 = 0, p1 = 1, p0 = 2;
    const float inv_g = 1.0f / GAMMA_;
    float res = 0.f;
    const float* pb = pred + (size_t)b * T_ * D_;

    for (int k = 2; k <= T_ + S_; ++k) {
        const int i = k - j;                 // DP row for this thread
        float val = BIGV;                    // boundary value for i==0 slots
        if (i >= 1 && i <= T_) {
            const float* pr = pb + (size_t)(i - 1) * D_;
            float sqP = 0.f, dot = 0.f;
            #pragma unroll
            for (int d = 0; d < D_; ++d) {
                const float pv = pr[d];
                sqP += pv * pv;
                dot += pv * tg[d];
            }
            const float cost = sqP + sqT - 2.f * dot;
            const float up = diag[p1][j];
            const float lf = diag[p1][j - 1];
            const float dg = diag[p2][j - 1];
            const float za = -up * inv_g, zb = -lf * inv_g, zc = -dg * inv_g;
            const float m  = fmaxf(fmaxf(za, zb), zc);
            const float sm = -GAMMA_ * (m + __logf(__expf(za - m) + __expf(zb - m) + __expf(zc - m)));
            val = cost + sm;
            if (i == T_ && j == S_) res = val;   // R[T][S]
        }
        diag[p0][j] = val;
        if (tid == 0) diag[p0][0] = BIGV;        // column-0 boundary
        __syncthreads();
        const int t = p2; p2 = p1; p1 = p0; p0 = t;
    }
    if (tid == S_ - 1) out_per_b[b] = res;
}

// -------------------------------------------------------------------------
// Kernel 4: deterministic fixed-order final reduction (no float atomics).
// -------------------------------------------------------------------------
__global__ void k_final(const float* __restrict__ ctc_pb,
                        const float* __restrict__ sdtw_pb,
                        float* __restrict__ out) {
    if (threadIdx.x == 0 && blockIdx.x == 0) {
        float sc = 0.f, ss = 0.f;
        for (int b = 0; b < B_; ++b) { sc += ctc_pb[b]; ss += sdtw_pb[b]; }
        out[0] = sc / (float)B_ + ss / (float)B_;
    }
}

extern "C" void kernel_launch(void* const* d_in, const int* in_sizes, int n_in,
                              void* d_out, int out_size, void* d_ws, size_t ws_size,
                              hipStream_t stream) {
    const float* lp      = (const float*)d_in[0];   // (B,T,V) log_probs
    const float* fm      = (const float*)d_in[1];   // (V,D)   feature_matrix
    const int*   targets = (const int*)d_in[2];     // (B,S)
    const int*   in_len  = (const int*)d_in[3];     // (B,)
    const int*   tgt_len = (const int*)d_in[4];     // (B,)

    float* pred    = (float*)d_ws;                  // B*T*D floats
    float* ctc_pb  = pred + (size_t)B_ * T_ * D_;   // B floats
    float* sdtw_pb = ctc_pb + B_;                   // B floats

    // GEMM: 4096 M-tiles of 16 rows, 8 waves/block -> 512 blocks
    k_gemm<<<(B_ * T_) / (16 * 8), 256, 0, stream>>>(lp, fm, pred);
    // CTC: one block per batch
    k_ctc<<<B_, 512, 0, stream>>>(lp, targets, in_len, tgt_len, ctc_pb);
    // soft-DTW: one block per batch (depends on pred, stream-ordered)
    k_sdtw<<<B_, 256, 0, stream>>>(pred, fm, targets, sdtw_pb);
    // final scalar
    k_final<<<1, 32, 0, stream>>>(ctc_pb, sdtw_pb, (float*)d_out);
}
